// ComplexMDTA_36928128811407
// MI455X (gfx1250) — matrix operand order repaired and checked by reading, not compiled, not run on hardware
//
#include <hip/hip_runtime.h>
#include <hip/hip_bf16.h>

typedef __attribute__((ext_vector_type(16))) __bf16   v16bf;
typedef __attribute__((ext_vector_type(8)))  float    v8f;
typedef __attribute__((ext_vector_type(4)))  unsigned v4u;
typedef __attribute__((ext_vector_type(8)))  int      v8i;
typedef __attribute__((ext_vector_type(4)))  int      v4i;

#define N_PIX 65536          // H*W
#define IMG_W 256
#define BATCH 4
#define C_IN  64
#define C3    192            // 3*C
#define NHEAD 8
#define EPSF  1e-6f

#if __has_builtin(__builtin_amdgcn_tensor_load_to_lds) && \
    __has_builtin(__builtin_amdgcn_s_wait_tensorcnt)
#define HAVE_TDM 1
#endif

// ---------- helpers ----------
static __device__ __forceinline__ unsigned bf_rne(float f) {
  unsigned u = __builtin_bit_cast(unsigned, f);
  u += 0x7FFFu + ((u >> 16) & 1u);
  return u >> 16;
}
static __device__ __forceinline__ __bf16 f2bf(float f) {
  unsigned short h = (unsigned short)bf_rne(f);
  return __builtin_bit_cast(__bf16, h);
}
static __device__ __forceinline__ float bf2f(unsigned short h) {
  unsigned u = ((unsigned)h) << 16;
  return __builtin_bit_cast(float, u);
}
static __device__ __forceinline__ unsigned pack_bf2(float a, float b) {
  return bf_rne(a) | (bf_rne(b) << 16);
}
static __device__ __forceinline__ v8f wmma_bf16(v16bf a, v16bf b, v8f c) {
  return __builtin_amdgcn_wmma_f32_16x16x32_bf16(false, a, false, b, (short)0, c,
                                                 false, false);
}

// A fragment from pixel-major bf16 activations (B*N, 64): two b128 loads.
static __device__ __forceinline__ v16bf load_a_packed(const __bf16* __restrict__ xt,
                                                      int mtile, int koff, int lane) {
  int p = mtile * 16 + (lane & 15);
  int kb = koff + ((lane & 16) ? 8 : 0);
  const __bf16* ptr = xt + (long)p * C_IN + kb;
  union { v16bf bf; v4u u[2]; } r;
  r.u[0] = *(const v4u*)(ptr);
  r.u[1] = *(const v4u*)(ptr + 16);
  return r.bf;
}

// B fragment from bf16 weights (OC, IC): two b128 loads at k offsets 8h and 16+8h.
static __device__ __forceinline__ v16bf load_b_packed(const __bf16* __restrict__ w,
                                                      int o0, int koff, int lane) {
  int o = o0 + (lane & 15);
  int kb = koff + ((lane & 16) ? 8 : 0);
  const __bf16* ptr = w + (long)o * C_IN + kb;
  union { v16bf bf; v4u u[2]; } r;
  r.u[0] = *(const v4u*)(ptr);
  r.u[1] = *(const v4u*)(ptr + 16);
  return r.bf;
}

// ---------- pre-pass: (B,C,N) f32 -> (B*N, C) bf16 transpose ----------
__global__ void transpose_to_bf16(const float* __restrict__ x, __bf16* __restrict__ xt) {
  int blk = blockIdx.x;                 // BATCH * (N_PIX/64) blocks
  int b  = blk >> 10;                   // N_PIX/64 = 1024
  int n0 = (blk & 1023) * 64;
  __shared__ float tile[64][65];
  long ibase = (long)b * C_IN * N_PIX + n0;
#pragma unroll
  for (int it = 0; it < 16; ++it) {
    int idx = it * 256 + threadIdx.x;
    int c = idx >> 6, n = idx & 63;
    tile[c][n] = x[ibase + (long)c * N_PIX + n];
  }
  __syncthreads();
  long obase = ((long)b * N_PIX + n0) * C_IN;   // element offset (even)
  unsigned* out32 = (unsigned*)xt;
#pragma unroll
  for (int it = 0; it < 8; ++it) {
    int idx = it * 256 + threadIdx.x;           // 2048 = 64 px * 32 chan-pairs
    int n = idx >> 5, c2 = (idx & 31) * 2;
    out32[(obase + (long)n * C_IN + c2) >> 1] = pack_bf2(tile[c2][n], tile[c2 + 1][n]);
  }
}

// ---------- pre-pass: weights f32 -> bf16 (+ optional negated copy) ----------
__global__ void w_to_bf16(const float* __restrict__ w, __bf16* __restrict__ out,
                          __bf16* __restrict__ outNeg, int n) {
  int i = blockIdx.x * 256 + threadIdx.x;
  if (i >= n) return;
  float v = w[i];
  out[i] = f2bf(v);
  if (outNeg) outNeg[i] = f2bf(-v);
}

__global__ void zero_f32(float* __restrict__ p, int n) {
  int i = blockIdx.x * 256 + threadIdx.x;
  if (i < n) p[i] = 0.f;
}

// ---------- stages 1 & 6: complex 1x1 conv == complex GEMM via WMMA ----------
template <int OUTMODE>   // 0: f32 out, 1: bf16 out (both channel-major)
__global__ void cgemm1x1_wmma(const __bf16* __restrict__ xtr, const __bf16* __restrict__ xti,
                              const __bf16* __restrict__ wr,  const __bf16* __restrict__ wi,
                              const __bf16* __restrict__ wiN,
                              void* __restrict__ outR, void* __restrict__ outI, int OC) {
  int lane = threadIdx.x & 31;
  int wid  = threadIdx.x >> 5;
  int m0   = (blockIdx.x * 4 + wid) * 4;   // 4 m-tiles per wave
  int o0   = blockIdx.y * 16;

  v16bf Br[2], Bi[2], BiN[2];
#pragma unroll
  for (int k = 0; k < 2; ++k) {
    Br[k]  = load_b_packed(wr,  o0, 32 * k, lane);
    Bi[k]  = load_b_packed(wi,  o0, 32 * k, lane);
    BiN[k] = load_b_packed(wiN, o0, 32 * k, lane);
  }

  int col  = lane & 15;
  int row0 = (lane & 16) ? 8 : 0;
#pragma unroll
  for (int t = 0; t < 4; ++t) {
    int mtile = m0 + t;
    v8f accR = {};
    v8f accI = {};
#pragma unroll
    for (int k = 0; k < 2; ++k) {
      v16bf Ar = load_a_packed(xtr, mtile, 32 * k, lane);
      v16bf Ai = load_a_packed(xti, mtile, 32 * k, lane);
      accR = wmma_bf16(Ar, Br[k],  accR);
      accR = wmma_bf16(Ai, BiN[k], accR);
      accI = wmma_bf16(Ai, Br[k],  accI);
      accI = wmma_bf16(Ar, Bi[k],  accI);
    }
    int pt = mtile * 16;
    int b  = pt >> 16;
    int nb = (pt & (N_PIX - 1)) + row0;
    long ofs = ((long)b * OC + (o0 + col)) * N_PIX + nb;
    if (OUTMODE == 0) {
      union { v8f f; v4u u[2]; } cr, ci;
      cr.f = accR; ci.f = accI;
      float* oR = (float*)outR; float* oI = (float*)outI;
      *(v4u*)(oR + ofs)     = cr.u[0];
      *(v4u*)(oR + ofs + 4) = cr.u[1];
      *(v4u*)(oI + ofs)     = ci.u[0];
      *(v4u*)(oI + ofs + 4) = ci.u[1];
    } else {
      v4u pr, pi;
      pr.x = pack_bf2(accR[0], accR[1]); pr.y = pack_bf2(accR[2], accR[3]);
      pr.z = pack_bf2(accR[4], accR[5]); pr.w = pack_bf2(accR[6], accR[7]);
      pi.x = pack_bf2(accI[0], accI[1]); pi.y = pack_bf2(accI[2], accI[3]);
      pi.z = pack_bf2(accI[4], accI[5]); pi.w = pack_bf2(accI[6], accI[7]);
      *(v4u*)((__bf16*)outR + ofs) = pr;
      *(v4u*)((__bf16*)outI + ofs) = pi;
    }
  }
}

// ---------- stage 2: complex depthwise 3x3, LDS-tiled (bf16 in/out) ----------
__global__ void dwconv3x3_complex(const __bf16* __restrict__ qr, const __bf16* __restrict__ qi,
                                  const float* __restrict__ wr, const float* __restrict__ wi,
                                  __bf16* __restrict__ outr, __bf16* __restrict__ outi) {
  __shared__ __align__(16) unsigned short lr[18 * 256];
  __shared__ __align__(16) unsigned short li[18 * 256];
  int rb = blockIdx.x;                 // 16 row-bands
  int bo = blockIdx.y;                 // b*192 + o
  int o  = bo % C3;
  long base = (long)bo * N_PIX;
  const unsigned short* qru = (const unsigned short*)qr;
  const unsigned short* qiu = (const unsigned short*)qi;

  // stage halo tile: rows rb*16-1 .. rb*16+16 (18 rows), full width
  for (int chunk = threadIdx.x; chunk < 576; chunk += 256) {   // 18 rows * 32 chunks
    int row = chunk >> 5;
    int c8  = (chunk & 31) * 8;
    int grow = rb * 16 + row - 1;
    v4u vrr = {0u, 0u, 0u, 0u}, vii = {0u, 0u, 0u, 0u};
    if (grow >= 0 && grow < IMG_W) {
      long g = base + (long)grow * IMG_W + c8;
      vrr = *(const v4u*)(qru + g);
      vii = *(const v4u*)(qiu + g);
    }
    ((v4u*)lr)[chunk] = vrr;
    ((v4u*)li)[chunk] = vii;
  }
  __syncthreads();

  float wr9[9], wi9[9];
#pragma unroll
  for (int j = 0; j < 9; ++j) { wr9[j] = wr[o * 9 + j]; wi9[j] = wi[o * 9 + j]; }

  int col = threadIdx.x;               // 256 columns
#pragma unroll 4
  for (int k = 0; k < 16; ++k) {
    float s_rr = 0.f, s_ii = 0.f, s_ir = 0.f, s_ri = 0.f;
#pragma unroll
    for (int kh = 0; kh < 3; ++kh) {
      int lrow = (k + kh) * 256;       // lds row (k-1+kh)+1
#pragma unroll
      for (int kw = 0; kw < 3; ++kw) {
        int cc = col + kw - 1;
        if (cc < 0 || cc >= IMG_W) continue;
        float xr = bf2f(lr[lrow + cc]);
        float xi = bf2f(li[lrow + cc]);
        float a = wr9[kh * 3 + kw], bb = wi9[kh * 3 + kw];
        s_rr += xr * a;  s_ii += xi * bb;
        s_ir += xi * a;  s_ri += xr * bb;
      }
    }
    long oidx = base + (long)(rb * 16 + k) * IMG_W + col;
    ((unsigned short*)outr)[oidx] = (unsigned short)bf_rne(s_rr - s_ii);
    ((unsigned short*)outi)[oidx] = (unsigned short)bf_rne(s_ir + s_ri);
  }
}

// ---------- stage 3: q/k magnitude norms over N ----------
__global__ void norm_kernel(const __bf16* __restrict__ dr, const __bf16* __restrict__ di,
                            float* __restrict__ qnorm, float* __restrict__ knorm) {
  int row = blockIdx.x;          // 0..511 : [qk][b][hc]
  int qk = row >> 8;
  int r  = row & 255;
  int b  = r >> 6;
  int hc = r & 63;
  long base = ((long)b * C3 + qk * 64 + hc) * (long)N_PIX;
  const unsigned short* dru = (const unsigned short*)dr;
  const unsigned short* diu = (const unsigned short*)di;
  float sum = 0.f;
  for (int n = threadIdx.x; n < N_PIX; n += 256) {
    float vr = bf2f(dru[base + n]);
    float vi = bf2f(diu[base + n]);
    float m = vr * vr + vi * vi + EPSF;
    sum += m * m;
  }
  __shared__ float red[256];
  red[threadIdx.x] = sum;
  __syncthreads();
  for (int s = 128; s > 0; s >>= 1) {
    if (threadIdx.x < s) red[threadIdx.x] += red[threadIdx.x + s];
    __syncthreads();
  }
  if (threadIdx.x == 0) {
    float v = sqrtf(red[0]) + EPSF;
    (qk ? knorm : qnorm)[r] = v;
  }
}

// ---------- stage 4a: raw scores via WMMA ----------
__global__ void scores_wmma(const __bf16* __restrict__ dr, const __bf16* __restrict__ di,
                            float* __restrict__ scores) {
  int lane  = threadIdx.x & 31;
  int wid   = threadIdx.x >> 5;
  int wglob = blockIdx.x * 4 + wid;     // 2048 waves, 64 slices per (b,h)
  int bh    = wglob >> 6;
  int slice = wglob & 63;
  int b = bh >> 3, h = bh & 7;
  long n0 = (long)slice * 1024;

  int arow = lane & 15;
  const unsigned short* asrc = (const unsigned short*)((arow < 8) ? dr : di);
  long abase = ((long)b * C3 + h * 8 + (arow & 7)) * (long)N_PIX + n0 +
               ((lane & 16) ? 8 : 0);
  int bcol = lane & 15;
  const unsigned short* bsrc = (const unsigned short*)((bcol < 8) ? dr : di);
  long bbase = ((long)b * C3 + 64 + h * 8 + (bcol & 7)) * (long)N_PIX + n0 +
               ((lane & 16) ? 8 : 0);

  v8f acc = {};
#pragma unroll 4
  for (int nc = 0; nc < 1024; nc += 32) {
    union { v16bf bf; v4u u[2]; } A, Bm;
    const unsigned short* ap = asrc + abase + nc;
    A.u[0]  = *(const v4u*)ap;
    A.u[1]  = *(const v4u*)(ap + 16);
    const unsigned short* bp = bsrc + bbase + nc;
    Bm.u[0] = *(const v4u*)bp;
    Bm.u[1] = *(const v4u*)(bp + 16);
    acc = wmma_bf16(A.bf, Bm.bf, acc);
  }

  __shared__ float sblk[64];
  if (threadIdx.x < 64) sblk[threadIdx.x] = 0.f;
  __syncthreads();
  bool hiHalf = (lane & 16) != 0;
  if (!hiHalf && bcol < 8) {            // rows 0..7 (q_r) x cols 0..7 (k_r)
#pragma unroll
    for (int i = 0; i < 8; ++i) atomicAdd(&sblk[i * 8 + bcol], acc[i]);
  } else if (hiHalf && bcol >= 8) {     // rows 8..15 (q_i) x cols 8..15 (k_i)
#pragma unroll
    for (int i = 0; i < 8; ++i) atomicAdd(&sblk[i * 8 + (bcol - 8)], acc[i]);
  }
  __syncthreads();
  if (threadIdx.x < 64) atomicAdd(&scores[bh * 64 + threadIdx.x], sblk[threadIdx.x]);
}

// ---------- stage 4b: normalize + softmax (tiny) ----------
__global__ void softmax_kernel(const float* __restrict__ scores,
                               const float* __restrict__ qnorm, const float* __restrict__ knorm,
                               const float* __restrict__ temperature,
                               float* __restrict__ attn) {
  int t = threadIdx.x;                  // 256 = 32 bh * 8 rows
  int bh = t >> 3, c = t & 7;
  int b = bh >> 3, h = bh & 7;
  float qn = qnorm[b * 64 + h * 8 + c];
  float tp = temperature[h];
  float sc[8];
  float m = -3.4e38f;
#pragma unroll
  for (int d = 0; d < 8; ++d) {
    sc[d] = scores[bh * 64 + c * 8 + d] / (qn * knorm[b * 64 + h * 8 + d]) * tp;
    m = fmaxf(m, sc[d]);
  }
  float sum = 0.f;
#pragma unroll
  for (int d = 0; d < 8; ++d) { sc[d] = __expf(sc[d] - m); sum += sc[d]; }
  float inv = 1.f / sum;
#pragma unroll
  for (int d = 0; d < 8; ++d) attn[bh * 64 + c * 8 + d] = sc[d] * inv;
}

// ---------- stage 5: out = attn @ v, written pixel-major bf16 for proj GEMM ----------
__global__ void apply_attn(const __bf16* __restrict__ dr, const __bf16* __restrict__ di,
                           const float* __restrict__ attn,
                           __bf16* __restrict__ atr, __bf16* __restrict__ ati) {
  int bh = blockIdx.y;
  int b = bh >> 3, h = bh & 7;
  __shared__ __align__(16) float a[64];
#ifdef HAVE_TDM
  // Tensor Data Mover: DMA the 64-float attn tile into LDS (1-row 2-D descriptor).
  if (threadIdx.x < 32) {               // one wave issues; EXEC ignored by TDM
    unsigned lds_off = (unsigned)(unsigned long long)(const void*)a;  // ISA: LDS addr = VA[31:0]
    unsigned long long g = (unsigned long long)(const void*)(attn + bh * 64);
    v4u g0 = { 1u,                                   // count=1, user descriptor
               lds_off,                               // lds_addr
               (unsigned)g,                           // global_addr[31:0]
               (unsigned)((g >> 32) & 0x1FFFFFFu) | (2u << 30) };  // addr[56:32] | type=2
    v8i g1 = { (int)(2u << 16),        // data_size=4B, no mask/flags
               (int)(64u << 16),       // tensor_dim0 = 64
               (int)(1u << 16),        // tensor_dim1 = 1
               (int)(64u << 16),       // tile_dim0 = 64
               1,                      // tile_dim1 = 1
               64, 0, 0 };             // tensor_dim0_stride = 64
    v4i gz4 = { 0, 0, 0, 0 };
    v8i gz8 = { 0, 0, 0, 0, 0, 0, 0, 0 };
    __builtin_amdgcn_tensor_load_to_lds(g0, g1, gz4, gz4, gz8, 0);
    __builtin_amdgcn_s_wait_tensorcnt(0);
  }
  __syncthreads();
#else
  if (threadIdx.x < 64) a[threadIdx.x] = attn[bh * 64 + threadIdx.x];
  __syncthreads();
#endif
  int n = blockIdx.x * 256 + threadIdx.x;
  long vbase = ((long)b * C3 + 128 + h * 8) * (long)N_PIX + n;
  const unsigned short* dru = (const unsigned short*)dr;
  const unsigned short* diu = (const unsigned short*)di;
  float vr[8], vi[8];
#pragma unroll
  for (int d = 0; d < 8; ++d) {
    vr[d] = bf2f(dru[vbase + (long)d * N_PIX]);
    vi[d] = bf2f(diu[vbase + (long)d * N_PIX]);
  }
  float orv[8], oiv[8];
#pragma unroll
  for (int c = 0; c < 8; ++c) {
    float sr = 0.f, si = 0.f;
#pragma unroll
    for (int d = 0; d < 8; ++d) {
      sr += a[c * 8 + d] * vr[d];
      si += a[c * 8 + d] * vi[d];
    }
    orv[c] = sr; oiv[c] = si;
  }
  long p = (long)b * N_PIX + n;
  v4u pr, pi;
  pr.x = pack_bf2(orv[0], orv[1]); pr.y = pack_bf2(orv[2], orv[3]);
  pr.z = pack_bf2(orv[4], orv[5]); pr.w = pack_bf2(orv[6], orv[7]);
  pi.x = pack_bf2(oiv[0], oiv[1]); pi.y = pack_bf2(oiv[2], oiv[3]);
  pi.z = pack_bf2(oiv[4], oiv[5]); pi.w = pack_bf2(oiv[6], oiv[7]);
  *(v4u*)(atr + p * C_IN + h * 8) = pr;
  *(v4u*)(ati + p * C_IN + h * 8) = pi;
}

// ---------- host glue ----------
extern "C" void kernel_launch(void* const* d_in, const int* in_sizes, int n_in,
                              void* d_out, int out_size, void* d_ws, size_t ws_size,
                              hipStream_t stream) {
  (void)in_sizes; (void)n_in; (void)out_size; (void)ws_size;
  const float* x_r     = (const float*)d_in[0];
  const float* x_i     = (const float*)d_in[1];
  const float* qkv_wr  = (const float*)d_in[2];
  const float* qkv_wi  = (const float*)d_in[3];
  const float* dw_wr   = (const float*)d_in[4];
  const float* dw_wi   = (const float*)d_in[5];
  const float* proj_wr = (const float*)d_in[6];
  const float* proj_wi = (const float*)d_in[7];
  const float* temp    = (const float*)d_in[8];

  const size_t NQ = (size_t)BATCH * C3 * N_PIX;     // 50331648 elements
  const size_t NX = (size_t)BATCH * N_PIX * C_IN;   // 16777216 elements
  char* base = (char*)d_ws;
  size_t o = 0;
  __bf16* qkv_r = (__bf16*)(base + o); o += NQ * 2;
  __bf16* qkv_i = (__bf16*)(base + o); o += NQ * 2;
  __bf16* dw_r  = (__bf16*)(base + o); o += NQ * 2;
  __bf16* dw_i  = (__bf16*)(base + o); o += NQ * 2;
  float* scores = (float*)(base + o);  o += 2048 * 4;
  float* qn     = (float*)(base + o);  o += 256 * 4;
  float* kn     = (float*)(base + o);  o += 256 * 4;
  float* attn   = (float*)(base + o);  o += 2048 * 4;
  __bf16* xt_r  = (__bf16*)(base + o); o += NX * 2;
  __bf16* xt_i  = (__bf16*)(base + o); o += NX * 2;
  __bf16* wqr   = (__bf16*)(base + o); o += C3 * C_IN * 2;
  __bf16* wqi   = (__bf16*)(base + o); o += C3 * C_IN * 2;
  __bf16* wqiN  = (__bf16*)(base + o); o += C3 * C_IN * 2;
  __bf16* wpr   = (__bf16*)(base + o); o += C_IN * C_IN * 2;
  __bf16* wpi   = (__bf16*)(base + o); o += C_IN * C_IN * 2;
  __bf16* wpiN  = (__bf16*)(base + o); o += C_IN * C_IN * 2;
  __bf16* at_r  = xt_r;                             // reuse: xt dead after stage 1
  __bf16* at_i  = xt_i;

  float* out_r = (float*)d_out;
  float* out_i = out_r + (size_t)BATCH * C_IN * N_PIX;

  // pre-passes
  transpose_to_bf16<<<BATCH * (N_PIX / 64), 256, 0, stream>>>(x_r, xt_r);
  transpose_to_bf16<<<BATCH * (N_PIX / 64), 256, 0, stream>>>(x_i, xt_i);
  w_to_bf16<<<(C3 * C_IN + 255) / 256, 256, 0, stream>>>(qkv_wr, wqr, (__bf16*)nullptr, C3 * C_IN);
  w_to_bf16<<<(C3 * C_IN + 255) / 256, 256, 0, stream>>>(qkv_wi, wqi, wqiN, C3 * C_IN);
  w_to_bf16<<<(C_IN * C_IN + 255) / 256, 256, 0, stream>>>(proj_wr, wpr, (__bf16*)nullptr, C_IN * C_IN);
  w_to_bf16<<<(C_IN * C_IN + 255) / 256, 256, 0, stream>>>(proj_wi, wpi, wpiN, C_IN * C_IN);

  const int mtiles = (BATCH * N_PIX) / 16;          // 16384
  // stage 1: qkv complex 1x1 (WMMA) -> bf16 channel-major
  cgemm1x1_wmma<1><<<dim3(mtiles / 16, C3 / 16), 128, 0, stream>>>(
      xt_r, xt_i, wqr, wqi, wqiN, (void*)qkv_r, (void*)qkv_i, C3);
  // stage 2: complex depthwise 3x3 (LDS-tiled, bf16 -> bf16)
  dwconv3x3_complex<<<dim3(16, BATCH * C3), 256, 0, stream>>>(
      qkv_r, qkv_i, dw_wr, dw_wi, dw_r, dw_i);
  // stage 3: q/k norms
  norm_kernel<<<512, 256, 0, stream>>>(dw_r, dw_i, qn, kn);
  // stage 4: raw scores (WMMA) + softmax
  zero_f32<<<8, 256, 0, stream>>>(scores, 2048);
  scores_wmma<<<512, 128, 0, stream>>>(dw_r, dw_i, scores);
  softmax_kernel<<<1, 256, 0, stream>>>(scores, qn, kn, temp, attn);
  // stage 5: attention apply -> pixel-major bf16
  apply_attn<<<dim3(N_PIX / 256, BATCH * NHEAD), 256, 0, stream>>>(
      dw_r, dw_i, attn, at_r, at_i);
  // stage 6: proj complex 1x1 (WMMA) -> d_out (stacked r then i, f32)
  cgemm1x1_wmma<0><<<dim3(mtiles / 16, C_IN / 16), 128, 0, stream>>>(
      at_r, at_i, wpr, wpi, wpiN, (void*)out_r, (void*)out_i, C_IN);
}
